// LRSelfAttentionHead_35527969473122
// MI455X (gfx1250) — matrix operand order repaired and checked by reading, not compiled, not run on hardware
//
#include <hip/hip_runtime.h>
#include <hip/hip_bf16.h>
#include <math.h>

// ---------- problem constants ----------
#define DIMC   1024
#define RANKC  64
#define BC     4
#define TC     4096
#define BT     (BC * TC)          // 16384 rows
#define SCALE  0.03125f           // 1/sqrt(1024)

// ---------- vector types ----------
typedef __bf16  v16bf __attribute__((ext_vector_type(16)));
typedef __bf16  v8bf  __attribute__((ext_vector_type(8)));
typedef float   v8f   __attribute__((ext_vector_type(8)));
typedef float   f32x8 __attribute__((ext_vector_type(8)));

static __device__ __forceinline__ v8f zero8() {
    v8f z;
#pragma unroll
    for (int i = 0; i < 8; ++i) z[i] = 0.0f;
    return z;
}

// WMMA wrapper: D = A(16x32 bf16) * B(32x16 bf16) + C(16x16 f32)
static __device__ __forceinline__ v8f wmma_bf16(v16bf a, v16bf b, v8f c) {
    return __builtin_amdgcn_wmma_f32_16x16x32_bf16(
        /*neg_a=*/false, a, /*neg_b=*/false, b,
        /*c_mod=*/(short)0, c, /*reuse_a=*/false, /*reuse_b=*/false);
}

// A operand (16x32, bf16 source). rowbase = start of this lane's matrix row.
// halves 0..7  <- K = k0 + 8g + h ; halves 8..15 <- K = k0 + 16 + 8g + h
static __device__ __forceinline__ v16bf load_A_bf16(const __bf16* rowbase, int k0, int g) {
    v8bf lo = *(const v8bf*)(rowbase + k0 + 8 * g);
    v8bf hi = *(const v8bf*)(rowbase + k0 + 16 + 8 * g);
    return __builtin_shufflevector(lo, hi, 0,1,2,3,4,5,6,7,8,9,10,11,12,13,14,15);
}

// A operand built from fp32 source (convert on the fly).
static __device__ __forceinline__ v16bf load_A_f32(const float* rowbase, int k0, int g) {
    f32x8 lo = *(const f32x8*)(rowbase + k0 + 8 * g);
    f32x8 hi = *(const f32x8*)(rowbase + k0 + 16 + 8 * g);
    v16bf a;
#pragma unroll
    for (int h = 0; h < 8; ++h) { a[h] = (__bf16)lo[h]; a[h + 8] = (__bf16)hi[h]; }
    return a;
}

// B operand (32x16). rowbase = this lane's column-row in the contraction-
// contiguous (transposed) B source. halves 0..7 <- K = k0 + 8g + h ; halves 8..15 <- K = k0 + 16 + 8g + h.
static __device__ __forceinline__ v16bf load_B(const __bf16* rowbase, int k0, int g) {
    v8bf lo = *(const v8bf*)(rowbase + k0 + 8 * g);
    v8bf hi = *(const v8bf*)(rowbase + k0 + 16 + 8 * g);
    return __builtin_shufflevector(lo, hi, 0,1,2,3,4,5,6,7,8,9,10,11,12,13,14,15);
}

// ---------- CDNA5 async global->LDS copy (ASYNCcnt path) ----------
// Low 32 bits of a generic shared-memory address ARE the LDS byte offset
// (ISA 10.2).
static __device__ __forceinline__ unsigned lds_off(const void* p) {
    return (unsigned)(unsigned long long)(uintptr_t)p;
}
static __device__ __forceinline__ void async_copy_b128(unsigned dst_lds, const void* gsrc) {
    asm volatile("global_load_async_to_lds_b128 %0, %1, off"
                 :: "v"(dst_lds), "v"(gsrc) : "memory");
}
static __device__ __forceinline__ void wait_async_0() { asm volatile("s_wait_asynccnt 0x0" ::: "memory"); }
static __device__ __forceinline__ void wait_async_2() { asm volatile("s_wait_asynccnt 0x2" ::: "memory"); }
static __device__ __forceinline__ void wait_async_4() { asm volatile("s_wait_asynccnt 0x4" ::: "memory"); }
static __device__ __forceinline__ void wait_async_6() { asm volatile("s_wait_asynccnt 0x6" ::: "memory"); }

// ============================================================
// K0: convert Qb = bf16(Q/32), Kb = bf16(K), VOt[e][d] = bf16(VO[d][e])
// ============================================================
__global__ __launch_bounds__(256) void k_convert(const float* __restrict__ Q,
                                                 const float* __restrict__ K,
                                                 const float* __restrict__ VO,
                                                 __bf16* __restrict__ Qb,
                                                 __bf16* __restrict__ Kb,
                                                 __bf16* __restrict__ VOt) {
    const int NQ = RANKC * DIMC;                       // 65536
    const int NTOT = 2 * NQ + DIMC * DIMC;             // + 1048576
    for (int idx = blockIdx.x * blockDim.x + threadIdx.x; idx < NTOT;
         idx += gridDim.x * blockDim.x) {
        if (idx < NQ) {
            Qb[idx] = (__bf16)(Q[idx] * SCALE);
        } else if (idx < 2 * NQ) {
            int j = idx - NQ;
            Kb[j] = (__bf16)(K[j]);
        } else {
            int j = idx - 2 * NQ;
            int e = j >> 10, d = j & 1023;
            VOt[j] = (__bf16)(VO[d * DIMC + e]);       // VOt[e][d]
        }
    }
}

// ============================================================
// K1: projections. xq = (x @ (Q/32)^T), xk = x @ K^T, stored bf16 row-major.
// Async double-buffered staging of x (fp32) + Qb/Kb tiles per 32-deep d-step.
// ============================================================
__global__ __launch_bounds__(256) void k_proj(const float* __restrict__ x,
                                              const __bf16* __restrict__ Qb,
                                              const __bf16* __restrict__ Kb,
                                              __bf16* __restrict__ xq,
                                              __bf16* __restrict__ xk) {
    __shared__ __attribute__((aligned(32))) float  xl[2][64 * 32];   // 16 KB
    __shared__ __attribute__((aligned(32))) __bf16 ql[2][64 * 32];   //  8 KB
    __shared__ __attribute__((aligned(32))) __bf16 kl[2][64 * 32];   //  8 KB

    const int tid = threadIdx.x;
    const int lane = tid & 31, g = lane >> 4, n = lane & 15;
    const int w = tid >> 5, rg = w & 3;
    const long R0 = (long)blockIdx.x * 64;

    v8f acc[4];
#pragma unroll
    for (int c = 0; c < 4; ++c) acc[c] = zero8();

    auto issue_tiles = [&](int bb, int d0) {
        // x tile: 64 rows x 32 f32 = 512 x 16B chunks (2/thread)
#pragma unroll
        for (int r = 0; r < 2; ++r) {
            int chunk = tid + r * 256;
            int row = chunk >> 3, seg = chunk & 7;
            async_copy_b128(lds_off(&xl[bb][row * 32 + seg * 4]),
                            x + (R0 + row) * DIMC + d0 + seg * 4);
        }
        // Qb / Kb tiles: 64 rows x 32 bf16 = 256 x 16B chunks each (1/thread)
        {
            int row = tid >> 2, seg = tid & 3;
            async_copy_b128(lds_off(&ql[bb][row * 32 + seg * 8]),
                            Qb + row * DIMC + d0 + seg * 8);
            async_copy_b128(lds_off(&kl[bb][row * 32 + seg * 8]),
                            Kb + row * DIMC + d0 + seg * 8);
        }
    };

    issue_tiles(0, 0);
    for (int it = 0; it < DIMC / 32; ++it) {
        const int cur = it & 1;
        const bool has_next = (it + 1 < DIMC / 32);
        if (has_next) issue_tiles(cur ^ 1, (it + 1) * 32);
        if (has_next) wait_async_4(); else wait_async_0();
        __syncthreads();

        const __bf16* Wl = (w < 4) ? &ql[cur][0] : &kl[cur][0];
        v16bf a = load_A_f32(&xl[cur][(rg * 16 + n) * 32], 0, g);
#pragma unroll
        for (int c = 0; c < 4; ++c)
            acc[c] = wmma_bf16(a, load_B(Wl + (c * 16 + n) * 32, 0, g), acc[c]);
        __syncthreads();
    }

    __bf16* dst = (w < 4) ? xq : xk;
#pragma unroll
    for (int c = 0; c < 4; ++c)
#pragma unroll
        for (int i = 0; i < 8; ++i)
            dst[(R0 + rg * 16 + i + 8 * g) * RANKC + c * 16 + n] = (__bf16)acc[c][i];
}

// ============================================================
// K2: v = x @ VO, stored TRANSPOSED: vT[b][e][u] (bf16).
// Async double-buffered staging of x (fp32) + VOt tiles per d-step.
// Block covers 64 rows x 256 cols.
// ============================================================
__global__ __launch_bounds__(256) void k_xvo(const float* __restrict__ x,
                                             const __bf16* __restrict__ VOt,
                                             __bf16* __restrict__ vT) {
    __shared__ __attribute__((aligned(32))) float  xl[2][64 * 32];    // 16 KB
    __shared__ __attribute__((aligned(32))) __bf16 vtl[2][256 * 32];  // 32 KB

    const int tid = threadIdx.x;
    const int lane = tid & 31, g = lane >> 4, n = lane & 15;
    const int w = tid >> 5, rg = w & 3;
    const int b = blockIdx.z, rt = blockIdx.y;
    const long R0 = (long)b * TC + rt * 64;
    const int cbase = blockIdx.x * 256;
    const int c0loc = (w >> 2) * 128;        // wave's local column base (0/128)

    v8f acc[8];
#pragma unroll
    for (int c = 0; c < 8; ++c) acc[c] = zero8();

    auto issue_tiles = [&](int bb, int d0) {
        // x tile: 512 chunks (2/thread)
#pragma unroll
        for (int r = 0; r < 2; ++r) {
            int chunk = tid + r * 256;
            int row = chunk >> 3, seg = chunk & 7;
            async_copy_b128(lds_off(&xl[bb][row * 32 + seg * 4]),
                            x + (R0 + row) * DIMC + d0 + seg * 4);
        }
        // VOt tile: 256 rows x 32 bf16 = 1024 chunks (4/thread)
#pragma unroll
        for (int r = 0; r < 4; ++r) {
            int chunk = tid + r * 256;
            int row = chunk >> 2, seg = chunk & 3;
            async_copy_b128(lds_off(&vtl[bb][row * 32 + seg * 8]),
                            VOt + (long)(cbase + row) * DIMC + d0 + seg * 8);
        }
    };

    issue_tiles(0, 0);
    for (int it = 0; it < DIMC / 32; ++it) {
        const int cur = it & 1;
        const bool has_next = (it + 1 < DIMC / 32);
        if (has_next) issue_tiles(cur ^ 1, (it + 1) * 32);
        if (has_next) wait_async_6(); else wait_async_0();
        __syncthreads();

        v16bf a = load_A_f32(&xl[cur][(rg * 16 + n) * 32], 0, g);
#pragma unroll
        for (int ct = 0; ct < 8; ++ct) {
            const __bf16* brow = &vtl[cur][(c0loc + ct * 16 + n) * 32];
            acc[ct] = wmma_bf16(a, load_B(brow, 0, g), acc[ct]);
        }
        __syncthreads();
    }
#pragma unroll
    for (int ct = 0; ct < 8; ++ct)
#pragma unroll
        for (int i = 0; i < 8; ++i)
            vT[((long)b * DIMC + cbase + c0loc + ct * 16 + n) * TC +
               rt * 64 + rg * 16 + i + 8 * g] = (__bf16)acc[ct][i];
}

// ============================================================
// K3: softmax stats (row max m, row sumexp l) via WMMA scores + online
// (m,l) update + shfl_xor lane reduction + cross-wave LDS merge.
// xk tiles async double-buffered through LDS.
// ============================================================
__global__ __launch_bounds__(256) void k_stats(const __bf16* __restrict__ xq,
                                               const __bf16* __restrict__ xk,
                                               float* __restrict__ smax,
                                               float* __restrict__ ssum) {
    __shared__ __attribute__((aligned(32))) __bf16 kl[2][64 * 64];   // 16 KB
    __shared__ float sM[2][64];
    __shared__ float sL[2][64];

    const int tid = threadIdx.x;
    const int lane = tid & 31, g = lane >> 4, n = lane & 15;
    const int w = tid >> 5, rg = w & 3, ch = w >> 2;
    const int b = blockIdx.y, rt = blockIdx.x;
    const long R0 = (long)b * TC + rt * 64;

    const __bf16* aq = xq + (R0 + rg * 16 + n) * RANKC;
    v16bf a0 = load_A_bf16(aq, 0, g);
    v16bf a1 = load_A_bf16(aq, 32, g);

    float m[8], l[8];
#pragma unroll
    for (int i = 0; i < 8; ++i) { m[i] = -3.0e38f; l[i] = 0.0f; }

    auto issue_tiles = [&](int bb, int u0) {
        // xk tile: 64 rows x 64 bf16 = 512 chunks (2/thread)
#pragma unroll
        for (int r = 0; r < 2; ++r) {
            int chunk = tid + r * 256;
            int row = chunk >> 3, seg = chunk & 7;
            async_copy_b128(lds_off(&kl[bb][row * 64 + seg * 8]),
                            xk + ((long)b * TC + u0 + row) * RANKC + seg * 8);
        }
    };

    issue_tiles(0, 0);
    for (int it = 0; it < TC / 64; ++it) {
        const int cur = it & 1;
        const bool has_next = (it + 1 < TC / 64);
        if (has_next) issue_tiles(cur ^ 1, (it + 1) * 64);
        if (has_next) wait_async_2(); else wait_async_0();
        __syncthreads();

        v8f s[2];
#pragma unroll
        for (int j = 0; j < 2; ++j) {
            const int sct = ch * 2 + j;
            const __bf16* brow = &kl[cur][(sct * 16 + n) * 64];
            v8f sa = zero8();
            sa = wmma_bf16(a0, load_B(brow, 0, g), sa);
            sa = wmma_bf16(a1, load_B(brow, 32, g), sa);
            s[j] = sa;
        }
#pragma unroll
        for (int i = 0; i < 8; ++i) {
            float s0 = s[0][i], s1 = s[1][i];
            float mx = fmaxf(m[i], fmaxf(s0, s1));
            l[i] = l[i] * __expf(m[i] - mx) + __expf(s0 - mx) + __expf(s1 - mx);
            m[i] = mx;
        }
        __syncthreads();
    }
#pragma unroll
    for (int i = 0; i < 8; ++i) {
#pragma unroll
        for (int mask = 1; mask <= 8; mask <<= 1) {
            float om = __shfl_xor(m[i], mask, 32);
            float ol = __shfl_xor(l[i], mask, 32);
            float M = fmaxf(m[i], om);
            l[i] = l[i] * __expf(m[i] - M) + ol * __expf(om - M);
            m[i] = M;
        }
    }
    if (n == 0) {
#pragma unroll
        for (int i = 0; i < 8; ++i) {
            sM[ch][rg * 16 + i + 8 * g] = m[i];
            sL[ch][rg * 16 + i + 8 * g] = l[i];
        }
    }
    __syncthreads();
    if (threadIdx.x < 64) {
        int r = threadIdx.x;
        float m0 = sM[0][r], m1 = sM[1][r];
        float l0 = sL[0][r], l1 = sL[1][r];
        float M = fmaxf(m0, m1);
        float L = l0 * __expf(m0 - M) + l1 * __expf(m1 - M);
        smax[R0 + r] = M;
        ssum[R0 + r] = L;
    }
}

// ============================================================
// K4: out[t, c0:c0+128] = (1/l) * sum_u exp(s(t,u)-m) * v[u, cols].
// Async double-buffered xk + vT tiles in LDS; S via WMMA, exp -> P in LDS,
// then WMMA P @ v. grid (8 colblocks, 64 rowtiles, 4 batch).
// ============================================================
__global__ __launch_bounds__(256) void k_attnout(const __bf16* __restrict__ xq,
                                                 const __bf16* __restrict__ xk,
                                                 const __bf16* __restrict__ vT,
                                                 const float* __restrict__ smax,
                                                 const float* __restrict__ ssum,
                                                 float* __restrict__ out) {
    __shared__ __attribute__((aligned(32))) __bf16 Pl[64 * 64];        // 8 KB
    __shared__ __attribute__((aligned(32))) __bf16 vbuf[2][128 * 64];  // 32 KB
    __shared__ __attribute__((aligned(32))) __bf16 kbuf[2][64 * 64];   // 16 KB

    const int tid = threadIdx.x;
    const int lane = tid & 31, g = lane >> 4, n = lane & 15;
    const int w = tid >> 5, rg = w & 3, wh = w >> 2;
    const int b = blockIdx.z, rt = blockIdx.y, cb = blockIdx.x;
    const long R0 = (long)b * TC + rt * 64;
    const int c0loc = wh * 64;               // wave's column base inside vbuf

    const __bf16* aq = xq + (R0 + rg * 16 + n) * RANKC;
    v16bf a0 = load_A_bf16(aq, 0, g);
    v16bf a1 = load_A_bf16(aq, 32, g);

    float mrow[8], linv[8];
#pragma unroll
    for (int i = 0; i < 8; ++i) {
        long r = R0 + rg * 16 + i + 8 * g;
        mrow[i] = smax[r];
        linv[i] = 1.0f / ssum[r];
    }

    v8f acc[4];
#pragma unroll
    for (int c = 0; c < 4; ++c) acc[c] = zero8();

    auto issue_tiles = [&](int bb, int u0) {
        // vT tile: 128 cols x 64 u = 1024 x 16B chunks (4/thread)
#pragma unroll
        for (int r = 0; r < 4; ++r) {
            int chunk = tid + r * 256;
            int e_l = chunk >> 3, seg = chunk & 7;
            const __bf16* gsrc =
                vT + ((long)b * DIMC + cb * 128 + e_l) * TC + u0 + seg * 8;
            async_copy_b128(lds_off(&vbuf[bb][e_l * 64 + seg * 8]), gsrc);
        }
        // xk tile: 64 rows x 64 r = 512 x 16B chunks (2/thread)
#pragma unroll
        for (int r = 0; r < 2; ++r) {
            int chunk = tid + r * 256;
            int u_l = chunk >> 3, seg = chunk & 7;
            const __bf16* gsrc = xk + ((long)b * TC + u0 + u_l) * RANKC + seg * 8;
            async_copy_b128(lds_off(&kbuf[bb][u_l * 64 + seg * 8]), gsrc);
        }
    };

    issue_tiles(0, 0);
    for (int it = 0; it < TC / 64; ++it) {
        const int cur = it & 1;
        const bool has_next = (it + 1 < TC / 64);
        if (has_next) issue_tiles(cur ^ 1, (it + 1) * 64);
        if (has_next) wait_async_6(); else wait_async_0();
        __syncthreads();                       // buffer `cur` fully populated

        // ---- scores + exp -> P (each wave writes 2 tiles of P) ----
#pragma unroll
        for (int j = 0; j < 2; ++j) {
            const int sct = wh * 2 + j;
            const __bf16* brow = &kbuf[cur][(sct * 16 + n) * 64];
            v8f s = zero8();
            s = wmma_bf16(a0, load_B(brow, 0, g), s);
            s = wmma_bf16(a1, load_B(brow, 32, g), s);
#pragma unroll
            for (int i = 0; i < 8; ++i)
                Pl[(rg * 16 + i + 8 * g) * 64 + sct * 16 + n] =
                    (__bf16)__expf(s[i] - mrow[i]);
        }
        __syncthreads();                       // P visible

        // ---- PV: acc += P(16x64) @ v(64x16 per tile), all from LDS ----
        const __bf16* prow = Pl + (rg * 16 + n) * 64;
        v16bf pa0 = load_A_bf16(prow, 0, g);
        v16bf pa1 = load_A_bf16(prow, 32, g);
#pragma unroll
        for (int ct = 0; ct < 4; ++ct) {
            const __bf16* vrow = &vbuf[cur][(c0loc + ct * 16 + n) * 64];
            acc[ct] = wmma_bf16(pa0, load_B(vrow, 0, g), acc[ct]);
            acc[ct] = wmma_bf16(pa1, load_B(vrow, 32, g), acc[ct]);
        }
        __syncthreads();   // PV reads done before next iter refills cur^1/P
    }

#pragma unroll
    for (int ct = 0; ct < 4; ++ct)
#pragma unroll
        for (int i = 0; i < 8; ++i)
            out[(R0 + rg * 16 + i + 8 * g) * DIMC + cb * 128 + c0loc + ct * 16 + n] =
                acc[ct][i] * linv[i];
}

// ============================================================
// host launcher
// ============================================================
extern "C" void kernel_launch(void* const* d_in, const int* in_sizes, int n_in,
                              void* d_out, int out_size, void* d_ws, size_t ws_size,
                              hipStream_t stream) {
    const float* x  = (const float*)d_in[0];   // [4,4096,1024]
    const float* Q  = (const float*)d_in[1];   // [64,1024]
    const float* K  = (const float*)d_in[2];   // [64,1024]
    const float* VO = (const float*)d_in[3];   // [1024,1024]
    float* out = (float*)d_out;                // [4,4096,1024]

    char* ws = (char*)d_ws;
    size_t off = 0;
    __bf16* Qb  = (__bf16*)(ws + off); off += (size_t)RANKC * DIMC * 2;     // 128K
    __bf16* Kb  = (__bf16*)(ws + off); off += (size_t)RANKC * DIMC * 2;     // 128K
    __bf16* VOt = (__bf16*)(ws + off); off += (size_t)DIMC * DIMC * 2;      // 2M
    __bf16* xqb = (__bf16*)(ws + off); off += (size_t)BT * RANKC * 2;       // 2M
    __bf16* xkb = (__bf16*)(ws + off); off += (size_t)BT * RANKC * 2;       // 2M
    __bf16* vT  = (__bf16*)(ws + off); off += (size_t)BC * DIMC * TC * 2;   // 32M
    float*  sm  = (float*)(ws + off);  off += (size_t)BT * 4;               // 64K
    float*  sl  = (float*)(ws + off);  off += (size_t)BT * 4;               // 64K
    (void)ws_size; (void)in_sizes; (void)n_in; (void)out_size;

    k_convert<<<dim3(2048), dim3(256), 0, stream>>>(Q, K, VO, Qb, Kb, VOt);
    k_proj   <<<dim3(BT / 64), dim3(256), 0, stream>>>(x, Qb, Kb, xqb, xkb);
    k_xvo    <<<dim3(4, TC / 64, BC), dim3(256), 0, stream>>>(x, VOt, vT);
    k_stats  <<<dim3(TC / 64, BC), dim3(256), 0, stream>>>(xqb, xkb, sm, sl);
    k_attnout<<<dim3(8, TC / 64, BC), dim3(256), 0, stream>>>(xqb, xkb, vT, sm, sl, out);
}
